// PulseMoE_45878840656621
// MI455X (gfx1250) — matrix operand order repaired and checked by reading, not compiled, not run on hardware
//
#include <hip/hip_runtime.h>
#include <hip/hip_bf16.h>
#include <stdint.h>

// Problem constants (match reference)
#define N_TOK 8192
#define DIM   1024
#define FDIM  4096
#define NEXP  8
#define CAP   1280          // int(1.25 * 8192/8)
#define KSTEP 64            // staged K depth (2 WMMA k-steps per barrier pair)
#define LDS_PAD 72          // halves per LDS row (144B: 16B-aligned, conflict-free)

typedef __bf16 v16bf __attribute__((ext_vector_type(16)));
typedef float  v8f   __attribute__((ext_vector_type(8)));
typedef unsigned int v4u __attribute__((ext_vector_type(4)));

union FragAB { v4u u[2]; v16bf b; };

__device__ __forceinline__ unsigned short f2bf(float f) {
  union { float f; uint32_t u; } v; v.f = f;
  uint32_t u = v.u;
  return (unsigned short)((u + 0x7FFFu + ((u >> 16) & 1u)) >> 16);  // RNE
}
__device__ __forceinline__ uint32_t pk2bf(float lo, float hi) {
  return (uint32_t)f2bf(lo) | ((uint32_t)f2bf(hi) << 16);
}
__device__ __forceinline__ v8f vzero8() {
  v8f v;
#pragma unroll
  for (int i = 0; i < 8; ++i) v[i] = 0.f;
  return v;
}

// ---------------- init: zero y and aux accumulators ----------------
__global__ void k_init(float* __restrict__ y, float* __restrict__ imp) {
  size_t i = (size_t)blockIdx.x * blockDim.x + threadIdx.x;
  if (i * 4 < (size_t)N_TOK * DIM) {
    float4 z; z.x = z.y = z.z = z.w = 0.f;
    ((float4*)y)[i] = z;
  }
  if (i < NEXP) imp[i] = 0.f;
}

// ---------------- router: logits, softmax, top1, imp ----------------
__global__ void k_router(const float* __restrict__ x, const float* __restrict__ Wr,
                         const float* __restrict__ br, int* __restrict__ top1,
                         float* __restrict__ imp) {
  int lane = threadIdx.x & 31;
  int gw = blockIdx.x * (blockDim.x >> 5) + (threadIdx.x >> 5);  // global wave 0..255
  for (int it = 0; it < N_TOK / 256; ++it) {
    int t = gw + it * 256;
    float acc[NEXP];
#pragma unroll
    for (int e = 0; e < NEXP; ++e) acc[e] = 0.f;
    const float* xr = x + (size_t)t * DIM;
    for (int d0 = 0; d0 < DIM; d0 += 32) {
      float xv = xr[d0 + lane];
      const float4* w4 = (const float4*)(Wr + (size_t)(d0 + lane) * NEXP);
      float4 a = w4[0], b = w4[1];
      acc[0] += xv * a.x; acc[1] += xv * a.y; acc[2] += xv * a.z; acc[3] += xv * a.w;
      acc[4] += xv * b.x; acc[5] += xv * b.y; acc[6] += xv * b.z; acc[7] += xv * b.w;
    }
#pragma unroll
    for (int e = 0; e < NEXP; ++e) {
#pragma unroll
      for (int off = 16; off > 0; off >>= 1)
        acc[e] += __shfl_down(acc[e], off, 32);
    }
    if (lane == 0) {
      float lmax = -1e30f; int arg = 0;
#pragma unroll
      for (int e = 0; e < NEXP; ++e) {
        acc[e] += br[e];
        if (acc[e] > lmax) { lmax = acc[e]; arg = e; }
      }
      float p[NEXP], s = 0.f;
#pragma unroll
      for (int e = 0; e < NEXP; ++e) { p[e] = __expf(acc[e] - lmax); s += p[e]; }
      float inv = 1.f / s;
#pragma unroll
      for (int e = 0; e < NEXP; ++e) atomicAdd(&imp[e], p[e] * inv);
      top1[t] = arg;
    }
  }
}

// ---------------- rank: capacity cumsum -> slot2tok, load counts ----------------
__global__ void k_rank(const int* __restrict__ top1, int* __restrict__ s2t,
                       int* __restrict__ loadc) {
  __shared__ int sc[NEXP][256];
  int tid = threadIdx.x;
  for (int i = tid; i < NEXP * CAP; i += 256) s2t[i] = -1;

  int cnt[NEXP];
#pragma unroll
  for (int e = 0; e < NEXP; ++e) cnt[e] = 0;
  int base_tok = tid * 32;
  for (int i = 0; i < 32; ++i) {
    int tt = top1[base_tok + i];
#pragma unroll
    for (int e = 0; e < NEXP; ++e) cnt[e] += (tt == e) ? 1 : 0;
  }
#pragma unroll
  for (int e = 0; e < NEXP; ++e) sc[e][tid] = cnt[e];
  __syncthreads();
  for (int off = 1; off < 256; off <<= 1) {
    int tv[NEXP];
#pragma unroll
    for (int e = 0; e < NEXP; ++e) tv[e] = (tid >= off) ? sc[e][tid - off] : 0;
    __syncthreads();
#pragma unroll
    for (int e = 0; e < NEXP; ++e) sc[e][tid] += tv[e];
    __syncthreads();
  }
  int r[NEXP];
#pragma unroll
  for (int e = 0; e < NEXP; ++e) r[e] = sc[e][tid] - cnt[e];  // exclusive base
  for (int i = 0; i < 32; ++i) {
    int tok = base_tok + i;
    int tt = top1[tok];
    int rank = 0;
#pragma unroll
    for (int e = 0; e < NEXP; ++e)
      if (tt == e) { rank = r[e]; r[e] = rank + 1; }
    if (rank < CAP) s2t[tt * CAP + rank] = tok;
  }
  if (tid < NEXP) loadc[tid] = sc[tid][255];
}

// Fragment-load + 16x WMMA over a staged [128 x KSTEP] A tile and [KSTEP x 128]
// (transposed) B tile. Shared by both GEMMs.
__device__ __forceinline__ void mma_tiles(const unsigned short* As,
                                          const unsigned short* Bs,
                                          int lane, int waveM, int waveN,
                                          v8f acc[4][2]) {
#pragma unroll
  for (int s = 0; s < KSTEP / 32; ++s) {
    FragAB a[4], b[2];
    int aco = s * 64 + (lane >> 4) * 16;  // byte offset of first 8-half chunk
#pragma unroll
    for (int tm = 0; tm < 4; ++tm) {
      int m = waveM * 64 + tm * 16 + (lane & 15);
      const v4u* p = (const v4u*)((const char*)As + m * (LDS_PAD * 2) + aco);
      a[tm].u[0] = p[0];
      a[tm].u[1] = p[2];  // +32 bytes (K+16 chunk)
    }
    int bco = s * 64 + (lane >> 4) * 16;
#pragma unroll
    for (int tn = 0; tn < 2; ++tn) {
      int n = waveN * 32 + tn * 16 + (lane & 15);
      const v4u* p = (const v4u*)((const char*)Bs + n * (LDS_PAD * 2) + bco);
      b[tn].u[0] = p[0];
      b[tn].u[1] = p[2];  // +32 bytes (K+16 chunk)
    }
#pragma unroll
    for (int tm = 0; tm < 4; ++tm)
#pragma unroll
      for (int tn = 0; tn < 2; ++tn)
        acc[tm][tn] = __builtin_amdgcn_wmma_f32_16x16x32_bf16(
            false, a[tm].b, false, b[tn].b, (short)0, acc[tm][tn], false, false);
  }
}

// ---------------- GEMM1: hid_bits = spike(x_gather @ W1 - theta) ----------------
__global__ void __launch_bounds__(256)
k_gemm1(const float* __restrict__ x, const float* __restrict__ W1,
        const float* __restrict__ theta, const int* __restrict__ s2t,
        uint32_t* __restrict__ hid) {
  __shared__ __align__(16) unsigned short As[128 * LDS_PAD];
  __shared__ __align__(16) unsigned short Bs[128 * LDS_PAD];
  int tid = threadIdx.x;
  int lane = tid & 31;
  int wave = tid >> 5;
  int waveM = wave >> 2, waveN = wave & 3;
  int e = blockIdx.z;
  int mBase = blockIdx.y * 128;
  int n0 = blockIdx.x * 128;

  // A-staging role: 2 threads per row, 32 k-values each
  int tr = tid >> 1;
  int kh = (tid & 1) * 32;
  int tokA = s2t[e * CAP + mBase + tr];
  const float* aSrc = x + (size_t)(tokA < 0 ? 0 : tokA) * DIM + kh;
  // B-staging role: thread covers k-rows dr and dr+32, 16 f-values each
  int dr = tid >> 3;
  int fo = (tid & 7) * 16;
  const float* bSrc = W1 + ((size_t)e * DIM + dr) * FDIM + n0 + fo;

  float theta_e = theta[e];
  v8f acc[4][2];
#pragma unroll
  for (int i = 0; i < 4; ++i)
#pragma unroll
    for (int j = 0; j < 2; ++j) acc[i][j] = vzero8();

  uint32_t* AsU = (uint32_t*)As;
  for (int k0 = 0; k0 < DIM; k0 += KSTEP) {
    {  // stage A (fp32 -> bf16, row-major [128][KSTEP])
      union { float4 q[8]; float f[32]; } aa;
      if (tokA >= 0) {
        const float4* s = (const float4*)(aSrc + k0);
#pragma unroll
        for (int q = 0; q < 8; ++q) aa.q[q] = s[q];
      } else {
#pragma unroll
        for (int j = 0; j < 32; ++j) aa.f[j] = 0.f;
      }
      int ub = (tr * LDS_PAD + kh) >> 1;
#pragma unroll
      for (int j = 0; j < 16; ++j) AsU[ub + j] = pk2bf(aa.f[2 * j], aa.f[2 * j + 1]);
    }
#pragma unroll
    for (int half = 0; half < 2; ++half) {  // stage B transposed: Bs[n][k]
      int kr = dr + half * 32;
      const float4* s = (const float4*)(bSrc + (size_t)(k0 + half * 32) * FDIM);
      union { float4 q[4]; float f[16]; } bb;
      bb.q[0] = s[0]; bb.q[1] = s[1]; bb.q[2] = s[2]; bb.q[3] = s[3];
#pragma unroll
      for (int j = 0; j < 16; ++j) Bs[(fo + j) * LDS_PAD + kr] = f2bf(bb.f[j]);
    }
    if (k0 + KSTEP < DIM)
      __builtin_prefetch(bSrc + (size_t)(k0 + KSTEP) * FDIM, 0, 0);
    __syncthreads();
    mma_tiles(As, Bs, lane, waveM, waveN, acc);
    __syncthreads();
  }

  // epilogue: spike -> bit-pack via ballot; each wave owns one 32-col word
  int wcol = (n0 + waveN * 32) >> 5;
#pragma unroll
  for (int tm = 0; tm < 4; ++tm) {
#pragma unroll
    for (int j = 0; j < 8; ++j) {
      unsigned long long b0 = __ballot(acc[tm][0][j] >= theta_e);
      unsigned long long b1 = __ballot(acc[tm][1][j] >= theta_e);
      if (lane == 0) {
        uint32_t wlo = (uint32_t)((b0 & 0xFFFFull) | ((b1 & 0xFFFFull) << 16));
        uint32_t whi = (uint32_t)(((b0 >> 16) & 0xFFFFull) |
                                  (((b1 >> 16) & 0xFFFFull) << 16));
        int m = mBase + waveM * 64 + tm * 16 + j;
        hid[((size_t)e * CAP + m) * (FDIM / 32) + wcol] = wlo;
        hid[((size_t)e * CAP + m + 8) * (FDIM / 32) + wcol] = whi;
      }
    }
  }
}

// ---------------- GEMM2: y[token] = hid_bits @ W2 (scatter) ----------------
__global__ void __launch_bounds__(256)
k_gemm2(const uint32_t* __restrict__ hid, const float* __restrict__ W2,
        const int* __restrict__ s2t, float* __restrict__ y) {
  __shared__ __align__(16) unsigned short As[128 * LDS_PAD];
  __shared__ __align__(16) unsigned short Bs[128 * LDS_PAD];
  int tid = threadIdx.x;
  int lane = tid & 31;
  int wave = tid >> 5;
  int waveM = wave >> 2, waveN = wave & 3;
  int e = blockIdx.z;
  int mBase = blockIdx.y * 128;
  int n0 = blockIdx.x * 128;

  int dr = tid >> 3;
  int fo = (tid & 7) * 16;
  const float* bSrc = W2 + ((size_t)e * FDIM + dr) * DIM + n0 + fo;
  const uint32_t* aSrc = hid + ((size_t)e * CAP + mBase) * (FDIM / 32);

  v8f acc[4][2];
#pragma unroll
  for (int i = 0; i < 4; ++i)
#pragma unroll
    for (int j = 0; j < 2; ++j) acc[i][j] = vzero8();

  uint32_t* AsU = (uint32_t*)As;
  // A staging role: 2 threads per row, one 32-bit word (32 k) each
  int atr = tid >> 1;
  int aw = tid & 1;
  for (int kt = 0; kt < FDIM / KSTEP; ++kt) {
    {  // stage A: expand 32 bits -> 32 bf16 {0,1}
      uint32_t w = aSrc[(size_t)atr * (FDIM / 32) + kt * 2 + aw];
      int ub = atr * (LDS_PAD / 2) + aw * 16;
#pragma unroll
      for (int j = 0; j < 16; ++j) {
        uint32_t u = (((w >> (2 * j)) & 1u) ? 0x3F80u : 0u) |
                     (((w >> (2 * j + 1)) & 1u) ? 0x3F800000u : 0u);
        AsU[ub + j] = u;
      }
    }
#pragma unroll
    for (int half = 0; half < 2; ++half) {  // stage B transposed: Bs[n][k]
      int kr = dr + half * 32;
      const float4* s = (const float4*)(bSrc + (size_t)(kt * KSTEP + half * 32) * DIM);
      union { float4 q[4]; float f[16]; } bb;
      bb.q[0] = s[0]; bb.q[1] = s[1]; bb.q[2] = s[2]; bb.q[3] = s[3];
#pragma unroll
      for (int j = 0; j < 16; ++j) Bs[(fo + j) * LDS_PAD + kr] = f2bf(bb.f[j]);
    }
    if (kt + 1 < FDIM / KSTEP)
      __builtin_prefetch(bSrc + (size_t)(kt + 1) * KSTEP * DIM, 0, 0);
    __syncthreads();
    mma_tiles(As, Bs, lane, waveM, waveN, acc);
    __syncthreads();
  }

  // epilogue: scatter rows back to tokens (dropped slots stay zero in y)
#pragma unroll
  for (int tm = 0; tm < 4; ++tm) {
#pragma unroll
    for (int j = 0; j < 8; ++j) {
      int m = waveM * 64 + tm * 16 + j + 8 * (lane >> 4);
      int tok = s2t[e * CAP + mBase + m];
      if (tok >= 0) {
#pragma unroll
        for (int tn = 0; tn < 2; ++tn) {
          int col = n0 + waveN * 32 + tn * 16 + (lane & 15);
          y[(size_t)tok * DIM + col] = acc[tm][tn][j];
        }
      }
    }
  }
}

// ---------------- aux loss ----------------
__global__ void k_aux(const float* __restrict__ imp, const int* __restrict__ loadc,
                      float* __restrict__ out_aux) {
  if (threadIdx.x == 0 && blockIdx.x == 0) {
    float si = 0.f, sl = 0.f, L[NEXP];
    for (int e = 0; e < NEXP; ++e) { si += imp[e]; L[e] = (float)loadc[e]; sl += L[e]; }
    float s = 0.f;
    for (int e = 0; e < NEXP; ++e)
      s += (imp[e] / (si + 1e-6f)) * (L[e] / (sl + 1e-6f));
    *out_aux = (1.f - s * (float)NEXP) * 0.01f;
  }
}

extern "C" void kernel_launch(void* const* d_in, const int* in_sizes, int n_in,
                              void* d_out, int out_size, void* d_ws, size_t ws_size,
                              hipStream_t stream) {
  const float* x     = (const float*)d_in[0];
  const float* Wr    = (const float*)d_in[1];
  const float* br    = (const float*)d_in[2];
  const float* W1    = (const float*)d_in[3];
  const float* W2    = (const float*)d_in[4];
  const float* theta = (const float*)d_in[5];
  float* y = (float*)d_out;

  // workspace layout
  char* ws = (char*)d_ws;
  float*    imp   = (float*)ws;                      // 8 f32
  int*      loadc = (int*)(ws + 64);                 // 8 i32
  int*      top1  = (int*)(ws + 128);                // 8192 i32
  int*      s2t   = (int*)(ws + 128 + N_TOK * 4);    // E*CAP i32
  uint32_t* hid   = (uint32_t*)(ws + 131072);        // E*CAP*F/32 u32 (~5MB)

  k_init<<<(N_TOK * DIM) / (4 * 256), 256, 0, stream>>>(y, imp);
  k_router<<<32, 256, 0, stream>>>(x, Wr, br, top1, imp);
  k_rank<<<1, 256, 0, stream>>>(top1, s2t, loadc);
  k_gemm1<<<dim3(FDIM / 128, CAP / 128, NEXP), 256, 0, stream>>>(x, W1, theta, s2t, hid);
  k_gemm2<<<dim3(DIM / 128, CAP / 128, NEXP), 256, 0, stream>>>(hid, W2, s2t, y);
  k_aux<<<1, 1, 0, stream>>>(imp, loadc, y + (size_t)N_TOK * DIM);
}
